// LEGOLLM_53317724012845
// MI455X (gfx1250) — matrix operand order repaired and checked by reading, not compiled, not run on hardware
//
#include <hip/hip_runtime.h>
#include <math.h>

// ---------------- model constants ----------------
#define S_   2048
#define D_   1024
#define H_   16
#define DH_  64
#define V_   32000
#define NB_  3
#define LPB_ 2
#define THR_ (1.02f / 32000.0f)

// ---------------- WMMA types ----------------
typedef __attribute__((ext_vector_type(16))) __bf16    v16bf;
typedef __attribute__((ext_vector_type(8)))  float     v8f;
typedef __attribute__((ext_vector_type(8)))  unsigned  u32x8;

// pack two fp32 -> one dword holding {bf16(hi), bf16(lo)} via v_perm_b32
__device__ __forceinline__ unsigned pk2bf(float lo, float hi) {
    return __builtin_amdgcn_perm(__builtin_bit_cast(unsigned, hi),
                                 __builtin_bit_cast(unsigned, lo), 0x07060302u);
}
__device__ __forceinline__ unsigned short f2bfu(float f) {
    return (unsigned short)(__builtin_bit_cast(unsigned, f) >> 16);
}

// hardware transcendental tanh (CDNA5 V_TANH_F32); 2 v_nop per TRANS hazard rule
__device__ __forceinline__ float tanh_hw(float x) {
    float r;
    asm volatile("v_tanh_f32 %0, %1\n\tv_nop\n\tv_nop" : "=v"(r) : "v"(x));
    return r;
}

__device__ __forceinline__ float gelu_tanh(float x) {
    float x3 = x * x * x;
    return 0.5f * x * (1.0f + tanh_hw(0.7978845608028654f * (x + 0.044715f * x3)));
}

// A-operand packed-pair index (u32 units) for 16x32 bf16 frag, VGPR v
__device__ __forceinline__ int a_pidx(int v, int hi) {
    return hi * 8 + v;
}

// ================= GEMM: C = epi(A @ B [+R]), bf16 WMMA, double-buffered =================
// A: [M,K] row-major f32.  B: row-major [K,N] (BT=0) or [N,K] (BT=1).
// EPI: 0 none, 1 residual add, 2 gelu.  PRED: skip rows where exited[row]!=0.
#define BM 64
#define BN 64
#define BK 32

template<int EPI, bool BT, bool PRED>
__global__ __launch_bounds__(128)
void gemm_wmma_kernel(const float* __restrict__ A, const float* __restrict__ B,
                      const float* __restrict__ R, float* __restrict__ C,
                      const int* __restrict__ exited, int M, int N, int K)
{
    // ping-pong packed bf16 pair tiles: [buf][row][k-pair], stride 17 dwords = 68B
    __shared__ unsigned As_pk[2][BM][BK / 2 + 1];
    __shared__ unsigned Bs_pk[2][BN][BK / 2 + 1];

    const int t    = threadIdx.x;
    const int wave = t >> 5;
    const int lane = t & 31;
    const int l15  = lane & 15;
    const int hi   = lane >> 4;
    const int m0   = blockIdx.x * BM;
    const int n0   = blockIdx.y * BN;

    // register-staged global tile loads
    auto load_tiles = [&](int k0, float4* areg, float4* breg) {
#pragma unroll
        for (int i = 0; i < 4; ++i) {
            int idx = t + i * 128;              // 0..511 float4 slots
            int r   = idx >> 3;
            int c   = (idx & 7) << 2;
            areg[i] = *(const float4*)(A + (size_t)(m0 + r) * K + k0 + c);
        }
        if (BT) {
#pragma unroll
            for (int i = 0; i < 4; ++i) {
                int idx = t + i * 128;
                int n   = idx >> 3;
                int c   = (idx & 7) << 2;
                breg[i] = *(const float4*)(B + (size_t)(n0 + n) * K + k0 + c);
            }
        } else {
#pragma unroll
            for (int i = 0; i < 2; ++i) {
                int idx = t + i * 128;          // 0..255
                int kp  = idx >> 4;             // k-pair 0..15
                int c   = (idx & 15) << 2;      // n col
                breg[2 * i + 0] = *(const float4*)(B + (size_t)(k0 + 2 * kp)     * N + n0 + c);
                breg[2 * i + 1] = *(const float4*)(B + (size_t)(k0 + 2 * kp + 1) * N + n0 + c);
            }
        }
    };
    // convert + store staged tiles into LDS buffer `buf`
    auto store_tiles = [&](int buf, const float4* areg, const float4* breg) {
#pragma unroll
        for (int i = 0; i < 4; ++i) {
            int idx = t + i * 128;
            int r   = idx >> 3;
            int c   = (idx & 7) << 2;
            As_pk[buf][r][(c >> 1) + 0] = pk2bf(areg[i].x, areg[i].y);
            As_pk[buf][r][(c >> 1) + 1] = pk2bf(areg[i].z, areg[i].w);
        }
        if (BT) {
#pragma unroll
            for (int i = 0; i < 4; ++i) {
                int idx = t + i * 128;
                int n   = idx >> 3;
                int c   = (idx & 7) << 2;
                Bs_pk[buf][n][(c >> 1) + 0] = pk2bf(breg[i].x, breg[i].y);
                Bs_pk[buf][n][(c >> 1) + 1] = pk2bf(breg[i].z, breg[i].w);
            }
        } else {
#pragma unroll
            for (int i = 0; i < 2; ++i) {
                int idx = t + i * 128;
                int kp  = idx >> 4;
                int c   = (idx & 15) << 2;
                const float4 f0 = breg[2 * i + 0];
                const float4 f1 = breg[2 * i + 1];
                Bs_pk[buf][c + 0][kp] = pk2bf(f0.x, f1.x);
                Bs_pk[buf][c + 1][kp] = pk2bf(f0.y, f1.y);
                Bs_pk[buf][c + 2][kp] = pk2bf(f0.z, f1.z);
                Bs_pk[buf][c + 3][kp] = pk2bf(f0.w, f1.w);
            }
        }
    };

    v8f acc[4];
    const v8f vzero = {0.f, 0.f, 0.f, 0.f, 0.f, 0.f, 0.f, 0.f};
#pragma unroll
    for (int j = 0; j < 4; ++j) acc[j] = vzero;

    // ---- pipeline prologue: tile 0 into buffer 0 ----
    {
        float4 areg[4], breg[4];
        load_tiles(0, areg, breg);
        store_tiles(0, areg, breg);
    }
    __syncthreads();

    int cur = 0;
    for (int k0 = 0; k0 < K; k0 += BK) {
        const bool nxt = (k0 + BK) < K;
        // issue next tile's global loads before the matrix math
        float4 anreg[4], bnreg[4];
        if (nxt) {
            load_tiles(k0 + BK, anreg, bnreg);
            if (k0 + 2 * BK < K) {
                __builtin_prefetch(A + (size_t)(m0 + (t >> 1)) * K + k0 + 2 * BK, 0, 0);
                if (BT) __builtin_prefetch(B + (size_t)(n0 + (t >> 1)) * K + k0 + 2 * BK, 0, 0);
                else    __builtin_prefetch(B + (size_t)(k0 + 2 * BK + (t >> 2)) * N + n0, 0, 0);
            }
        }

        // ---- compute from current buffer ----
        const int arow = wave * 16 + l15;
        u32x8 au;
#pragma unroll
        for (int v = 0; v < 8; ++v) au[v] = As_pk[cur][arow][a_pidx(v, hi)];
        const v16bf afr = __builtin_bit_cast(v16bf, au);

#pragma unroll
        for (int j = 0; j < 4; ++j) {
            const int bcol = j * 16 + l15;
            u32x8 bu;
#pragma unroll
            for (int v = 0; v < 8; ++v) bu[v] = Bs_pk[cur][bcol][hi * 8 + v];
            const v16bf bfr = __builtin_bit_cast(v16bf, bu);
            acc[j] = __builtin_amdgcn_wmma_f32_16x16x32_bf16(
                false, afr, false, bfr, (short)0, acc[j], false, false);
        }

        // ---- fill the other buffer while WMMAs drain ----
        if (nxt) store_tiles(cur ^ 1, anreg, bnreg);
        __syncthreads();
        cur ^= 1;
    }

    // ---- epilogue ----
#pragma unroll
    for (int j = 0; j < 4; ++j) {
#pragma unroll
        for (int r = 0; r < 8; ++r) {
            int row = m0 + wave * 16 + r + hi * 8;
            int col = n0 + j * 16 + l15;
            if (PRED && exited[row]) continue;
            float val = acc[j][r];
            if (EPI == 1) val += R[(size_t)row * N + col];
            if (EPI == 2) val = gelu_tanh(val);
            C[(size_t)row * N + col] = val;
        }
    }
}

// ================= Flash attention (WMMA) =================
// qkv: [S, 3D] f32 (q|k|v).  out: [S, D] f32.  One wave = one 16-query tile of one head.
__global__ __launch_bounds__(128)
void attn_kernel(const float* __restrict__ qkv, float* __restrict__ out)
{
    // P round-trip buffer: row stride 36 shorts = 72B (dword aligned)
    __shared__ unsigned short plds[4][16][36];

    const int t    = threadIdx.x;
    const int w    = t >> 5;
    const int lane = t & 31;
    const int l15  = lane & 15;
    const int hi   = lane >> 4;
    const int h    = blockIdx.y;
    const int qt   = blockIdx.x * 4 + w;
    const int qbase = qt * 16;
    const float scale = 0.125f;              // 1/sqrt(64)

    const float* qp = qkv + (size_t)h * DH_;
    const float* kp = qkv + D_ + (size_t)h * DH_;
    const float* vp = qkv + 2 * D_ + (size_t)h * DH_;

    // Q fragments (two K-chunks of 32 over DH=64), pre-scaled, float4 loads + perm pack
    v16bf aq[2];
    const int qrow = qbase + l15;
#pragma unroll
    for (int c = 0; c < 2; ++c) {
        u32x8 qa;
#pragma unroll
        for (int g = 0; g < 4; ++g) {
            int vb = (g >> 1) * 4 + (g & 1) * 2;                       // 0,2,4,6
            int kk = c * 32 + hi * 16 + g * 4;
            const float4 f = *(const float4*)(qp + (size_t)qrow * (3 * D_) + kk);
            qa[vb + 0] = pk2bf(f.x * scale, f.y * scale);
            qa[vb + 1] = pk2bf(f.z * scale, f.w * scale);
        }
        aq[c] = __builtin_bit_cast(v16bf, qa);
    }

    const v8f vzero = {0.f, 0.f, 0.f, 0.f, 0.f, 0.f, 0.f, 0.f};
    v8f o[4];
    float mrun[8], lrun[8];
#pragma unroll
    for (int j = 0; j < 4; ++j) o[j] = vzero;
#pragma unroll
    for (int r = 0; r < 8; ++r) { mrun[r] = -1e30f; lrun[r] = 0.f; }

    const int nkt = (qbase + 47) >> 5;       // 32-key tiles covering keys <= qbase+15
    for (int kt = 0; kt < nkt; ++kt) {
        // ---- scores: S = Q @ K^T, two 16-key halves (float4 K loads) ----
        v8f s[2]; s[0] = vzero; s[1] = vzero;
#pragma unroll
        for (int j = 0; j < 2; ++j) {
            const int key = kt * 32 + j * 16 + l15;
#pragma unroll
            for (int c = 0; c < 2; ++c) {
                u32x8 bu;
#pragma unroll
                for (int v2 = 0; v2 < 4; ++v2) {
                    int kk = c * 32 + hi * 16 + 4 * v2;
                    const float4 f = *(const float4*)(kp + (size_t)key * (3 * D_) + kk);
                    bu[2 * v2 + 0] = pk2bf(f.x, f.y);
                    bu[2 * v2 + 1] = pk2bf(f.z, f.w);
                }
                const v16bf bk = __builtin_bit_cast(v16bf, bu);
                s[j] = __builtin_amdgcn_wmma_f32_16x16x32_bf16(
                    false, aq[c], false, bk, (short)0, s[j], false, false);
            }
        }
        // ---- causal mask (boundary tiles only) ----
        if (kt * 32 + 31 > qbase) {
#pragma unroll
            for (int j = 0; j < 2; ++j) {
                const int key = kt * 32 + j * 16 + l15;
#pragma unroll
                for (int r = 0; r < 8; ++r) {
                    int q = qbase + r + hi * 8;
                    if (key > q) s[j][r] = -1e30f;
                }
            }
        }
        // ---- online softmax (lanes<16: rows 0..7, lanes>=16: rows 8..15) ----
        float alpha[8];
#pragma unroll
        for (int r = 0; r < 8; ++r) {
            float mt = fmaxf(s[0][r], s[1][r]);
#pragma unroll
            for (int msk = 1; msk < 16; msk <<= 1) mt = fmaxf(mt, __shfl_xor(mt, msk, 16));
            float mnew = fmaxf(mrun[r], mt);
            alpha[r] = __expf(mrun[r] - mnew);
            mrun[r] = mnew;
            float p0 = __expf(s[0][r] - mnew);
            float p1 = __expf(s[1][r] - mnew);
            s[0][r] = p0; s[1][r] = p1;
            float rs = p0 + p1;
#pragma unroll
            for (int msk = 1; msk < 16; msk <<= 1) rs += __shfl_xor(rs, msk, 16);
            lrun[r] = lrun[r] * alpha[r] + rs;
        }
        // ---- P (C-layout) -> LDS (16-bit stores) ----
#pragma unroll
        for (int j = 0; j < 2; ++j)
#pragma unroll
            for (int r = 0; r < 8; ++r)
                plds[w][r + hi * 8][j * 16 + l15] = f2bfu(s[j][r]);
#pragma unroll
        for (int j = 0; j < 4; ++j)
#pragma unroll
            for (int r = 0; r < 8; ++r) o[j][r] *= alpha[r];

        // ---- read P back as A operand (8 dword LDS reads) ----
        const unsigned* prow = (const unsigned*)(&plds[w][l15][0]);
        u32x8 pu;
#pragma unroll
        for (int v = 0; v < 8; ++v) pu[v] = prow[a_pidx(v, hi)];
        const v16bf ap = __builtin_bit_cast(v16bf, pu);

        // ---- O += P @ V ----
#pragma unroll
        for (int j = 0; j < 4; ++j) {
            const int d = j * 16 + l15;
            u32x8 bu;
#pragma unroll
            for (int v = 0; v < 8; ++v) {
                int key = kt * 32 + hi * 16 + 2 * v;
                bu[v] = pk2bf(vp[(size_t)key * (3 * D_) + d],
                              vp[(size_t)(key + 1) * (3 * D_) + d]);
            }
            const v16bf bv = __builtin_bit_cast(v16bf, bu);
            o[j] = __builtin_amdgcn_wmma_f32_16x16x32_bf16(
                false, ap, false, bv, (short)0, o[j], false, false);
        }
    }

    // ---- normalize + store ----
#pragma unroll
    for (int r = 0; r < 8; ++r) {
        float inv = 1.0f / lrun[r];
        int srow = qbase + r + hi * 8;
#pragma unroll
        for (int j = 0; j < 4; ++j)
            out[(size_t)srow * D_ + h * DH_ + j * 16 + l15] = o[j][r] * inv;
    }
}

// ================= LayerNorm (gamma only) =================
__global__ __launch_bounds__(256)
void ln_kernel(const float* __restrict__ in, const float* __restrict__ g,
               float* __restrict__ out)
{
    __shared__ float red[256];
    const int s = blockIdx.x, t = threadIdx.x;
    const float* row = in + (size_t)s * D_;
    float v[4];
    float sum = 0.f;
#pragma unroll
    for (int i = 0; i < 4; ++i) { v[i] = row[t + 256 * i]; sum += v[i]; }
    red[t] = sum; __syncthreads();
    for (int o = 128; o > 0; o >>= 1) { if (t < o) red[t] += red[t + o]; __syncthreads(); }
    const float mu = red[0] * (1.0f / D_);
    __syncthreads();
    float sq = 0.f;
#pragma unroll
    for (int i = 0; i < 4; ++i) { float d = v[i] - mu; sq += d * d; }
    red[t] = sq; __syncthreads();
    for (int o = 128; o > 0; o >>= 1) { if (t < o) red[t] += red[t + o]; __syncthreads(); }
    const float rstd = rsqrtf(red[0] * (1.0f / D_) + 1e-5f);
#pragma unroll
    for (int i = 0; i < 4; ++i) {
        int d = t + 256 * i;
        out[(size_t)s * D_ + d] = (v[i] - mu) * rstd * g[d];
    }
}

// ================= confidence / early-exit =================
__global__ __launch_bounds__(256)
void conf_kernel(const float* __restrict__ logits, int* __restrict__ exited)
{
    const int s = blockIdx.x;
    if (exited[s]) return;                  // block-uniform
    __shared__ float red[256];
    const int t = threadIdx.x;
    const float* row = logits + (size_t)s * V_;
    float mx = -3.0e38f;
    for (int i = t; i < V_; i += 256) mx = fmaxf(mx, row[i]);
    red[t] = mx; __syncthreads();
    for (int o = 128; o > 0; o >>= 1) { if (t < o) red[t] = fmaxf(red[t], red[t + o]); __syncthreads(); }
    mx = red[0]; __syncthreads();
    float se = 0.f;
    for (int i = t; i < V_; i += 256) se += __expf(row[i] - mx);
    red[t] = se; __syncthreads();
    for (int o = 128; o > 0; o >>= 1) { if (t < o) red[t] += red[t + o]; __syncthreads(); }
    if (t == 0 && (1.0f / red[0]) > THR_) exited[s] = 1;
}

// ================= small utility kernels =================
__global__ void embed_kernel(const int* __restrict__ x, const float* __restrict__ emb,
                             float* __restrict__ h)
{
    int idx = blockIdx.x * 256 + threadIdx.x;
    int s = idx >> 10, d = idx & 1023;
    h[idx] = emb[(size_t)x[s] * D_ + d];
}

__global__ void select_kernel(float* __restrict__ h, const float* __restrict__ hout,
                              const int* __restrict__ exited)
{
    int idx = blockIdx.x * 256 + threadIdx.x;
    int s = idx >> 10;
    if (!exited[s]) h[idx] = hout[idx];
}

__global__ void zero_kernel(int* __restrict__ e)
{
    e[blockIdx.x * 256 + threadIdx.x] = 0;
}

// ================= host orchestration =================
extern "C" void kernel_launch(void* const* d_in, const int* in_sizes, int n_in,
                              void* d_out, int out_size, void* d_ws, size_t ws_size,
                              hipStream_t stream)
{
    (void)in_sizes; (void)n_in; (void)out_size; (void)ws_size;

    const int*   x    = (const int*)  d_in[0];
    const float* emb  = (const float*)d_in[1];
    const float* Wout = (const float*)d_in[2];
    const float* ln1  = (const float*)d_in[3];
    const float* Wqkv = (const float*)d_in[4];
    const float* Wo   = (const float*)d_in[5];
    const float* ln2  = (const float*)d_in[6];
    const float* W1   = (const float*)d_in[7];
    const float* W2   = (const float*)d_in[8];
    const float* lnf  = (const float*)d_in[9];
    float* out = (float*)d_out;

    char* ws = (char*)d_ws;
    float* h    = (float*)ws; ws += (size_t)S_ * D_ * 4;
    float* hout = (float*)ws; ws += (size_t)S_ * D_ * 4;
    float* xn   = (float*)ws; ws += (size_t)S_ * D_ * 4;
    float* qkv  = (float*)ws; ws += (size_t)S_ * 3 * D_ * 4;
    float* ao   = (float*)ws; ws += (size_t)S_ * D_ * 4;
    float* a1   = (float*)ws; ws += (size_t)S_ * 4 * D_ * 4;
    int*   exited = (int*)ws;

    zero_kernel<<<S_ / 256, 256, 0, stream>>>(exited);
    embed_kernel<<<(S_ * D_) / 256, 256, 0, stream>>>(x, emb, h);

    for (int b = 0; b < NB_; ++b) {
        hipMemcpyAsync(hout, h, (size_t)S_ * D_ * 4, hipMemcpyDeviceToDevice, stream);
        for (int l = b * LPB_; l < (b + 1) * LPB_; ++l) {
            ln_kernel<<<S_, 256, 0, stream>>>(hout, ln1 + (size_t)l * D_, xn);
            gemm_wmma_kernel<0, false, false><<<dim3(S_ / BM, 3 * D_ / BN), 128, 0, stream>>>(
                xn, Wqkv + (size_t)l * D_ * 3 * D_, nullptr, qkv, nullptr, S_, 3 * D_, D_);
            attn_kernel<<<dim3(S_ / 64, H_), 128, 0, stream>>>(qkv, ao);
            gemm_wmma_kernel<1, false, false><<<dim3(S_ / BM, D_ / BN), 128, 0, stream>>>(
                ao, Wo + (size_t)l * D_ * D_, hout, hout, nullptr, S_, D_, D_);
            ln_kernel<<<S_, 256, 0, stream>>>(hout, ln2 + (size_t)l * D_, xn);
            gemm_wmma_kernel<2, false, false><<<dim3(S_ / BM, 4 * D_ / BN), 128, 0, stream>>>(
                xn, W1 + (size_t)l * D_ * 4 * D_, nullptr, a1, nullptr, S_, 4 * D_, D_);
            gemm_wmma_kernel<1, false, false><<<dim3(S_ / BM, D_ / BN), 128, 0, stream>>>(
                a1, W2 + (size_t)l * 4 * D_ * D_, hout, hout, nullptr, S_, D_, 4 * D_);
        }
        // shared output head: write logits directly into d_out, only for live rows
        ln_kernel<<<S_, 256, 0, stream>>>(hout, lnf + (size_t)b * D_, xn);
        gemm_wmma_kernel<0, true, true><<<dim3(S_ / BM, V_ / BN), 128, 0, stream>>>(
            xn, Wout, nullptr, out, exited, S_, V_, D_);
        if (b < NB_ - 1) {
            conf_kernel<<<S_, 256, 0, stream>>>(out, exited);
            select_kernel<<<(S_ * D_) / 256, 256, 0, stream>>>(h, hout, exited);
        }
    }
}
